// ConvSparseCoding_16372415332583
// MI455X (gfx1250) — matrix operand order repaired and checked by reading, not compiled, not run on hardware
//
#include <hip/hip_runtime.h>

// ---- CDNA5 WMMA vector types ----
typedef __attribute__((ext_vector_type(16))) __bf16 v16bf;
typedef __attribute__((ext_vector_type(8)))  __bf16 v8bf;
typedef __attribute__((ext_vector_type(8)))  float  v8f;
typedef __attribute__((ext_vector_type(8)))  unsigned short u16x8;
typedef __attribute__((ext_vector_type(4)))  unsigned short u16x4;
typedef __attribute__((ext_vector_type(2)))  int    v2i;

// async global->LDS (gfx1250) if this toolchain exposes the builtin.
// Probe round 3: builtin exists, param0 = addrspace(1) v2i* (global src),
// param1 = LDS side, both non-const.
#if defined(__has_builtin)
# if __has_builtin(__builtin_amdgcn_global_load_async_to_lds_b64)
#  define HAVE_ASYNC_LDS 1
# endif
#endif
#ifdef HAVE_ASYNC_LDS
typedef __attribute__((address_space(1))) v2i* gas1_v2i;
typedef __attribute__((address_space(3))) v2i* gas3_v2i;
#endif

// ---- optimizer constants ----
#define RHO_      0.9f
#define LR_       0.001f
#define MOM_      0.9f
#define EPS_RMS   1e-8f
#define GAMMA_    0.1f
#define DIFF_EPS_ 1e-6f

constexpr int Bn = 16;   // batch
constexpr int Cn = 64;   // code channels
constexpr int Fn = 3;    // image channels
constexpr int Hn = 64;   // image rows
constexpr int Wn = 64;   // image cols
constexpr int Rn = 9;    // kernel
constexpr int Pn = 72;   // code rows
constexpr int Qn = 72;   // code cols

constexpr int KFWD_CH = (81 * Cn) / 32;  // 162 k-chunks
constexpr int KBWD    = Fn * 81;         // 243
constexpr int KBWD_CH = 8;               // padded to 256

// Internal state (x f32, x bf16 shadow, a1, a2) is NHWC:
//   idx = ((b*Pn + p)*Qn + q)*Cn + c

// =====================================================================
// Pack flipped weights into WMMA fragment order (bf16), once per call.
// =====================================================================
__global__ void pack_weights(const float* __restrict__ Wg,
                             __bf16* __restrict__ pf,
                             __bf16* __restrict__ pb) {
  int t = blockIdx.x * blockDim.x + threadIdx.x;

  const int nf = KFWD_CH * 32 * 16;         // 82944
  if (t < nf) {
    int idx  = t & 15;
    int lane = (t >> 4) & 31;
    int kc   = t >> 9;
    int n    = lane & 15;
    int k    = kc * 32 + ((lane >> 4) * 8) + (idx < 8 ? idx : 8 + idx);
    int uv = k >> 6, c = k & 63;
    int u = uv / 9, v = uv - u * 9;
    float val = 0.f;
    if (n < Fn)
      val = Wg[((n * Cn + c) * Rn + (8 - u)) * Rn + (8 - v)];
    pf[t] = (__bf16)val;
  }

  const int nb = 4 * KBWD_CH * 32 * 16;     // 16384
  if (t < nb) {
    int idx  = t & 15;
    int lane = (t >> 4) & 31;
    int kc   = (t >> 9) & 7;
    int cb   = t >> 12;
    int m    = lane & 15;
    int c    = cb * 16 + m;
    int kb   = (lane >> 4) * 16;
    int k    = kc * 32 + kb + idx;
    float val = 0.f;
    if (k < KBWD) {
      int f = k / 81, r1 = k - f * 81;
      int u = r1 / 9, v = r1 - u * 9;
      val = Wg[((f * Cn + c) * Rn + (8 - u)) * Rn + (8 - v)];
    }
    pb[t] = (__bf16)val;
  }
}

__global__ void init_state(float* __restrict__ x, float* __restrict__ a1,
                           float* __restrict__ a2, __bf16* __restrict__ xbf,
                           int n) {
  int t = blockIdx.x * blockDim.x + threadIdx.x;
  if (t < n) { x[t] = 0.f; a1[t] = 0.f; a2[t] = 0.f; xbf[t] = (__bf16)0.f; }
}

// =====================================================================
// Forward implicit GEMM.  Block = 4 waves = one (b, image row i).
// Per u: copy one bf16 x-row (9216 B, contiguous NHWC) into LDS —
// async global->LDS when available — then 18 WMMA chunks read A via
// ds_load_b128.  Residual stored as bf16 (it feeds a bf16 WMMA operand).
// =====================================================================
__global__ __launch_bounds__(128)
void fwd_residual(const __bf16* __restrict__ xbf, const float* __restrict__ inp,
                  const __bf16* __restrict__ pf, __bf16* __restrict__ ebf) {
  __shared__ __bf16 xrow[Qn * Cn];          // [col][c], 9216 B

  const int tid  = threadIdx.x;
  const int lane = tid & 31;
  const int wave = tid >> 5;
  const int b    = blockIdx.x >> 6;         // Hn = 64
  const int i    = blockIdx.x & 63;
  const int j0   = wave << 4;
  const int mrow = lane & 15;               // A row = pixel in strip
  const int kb   = (lane >> 4) * 8;         // A k-sub-base
  const int j    = j0 + mrow;

  v8f acc = {0.f, 0.f, 0.f, 0.f, 0.f, 0.f, 0.f, 0.f};

#pragma unroll 1
  for (int u = 0; u < Rn; ++u) {
    __syncthreads();
    const __bf16* srcb = xbf + (size_t)(b * Pn + (i + u)) * Qn * Cn;
#ifdef HAVE_ASYNC_LDS
#pragma unroll
    for (int s = 0; s < 9; ++s) {           // 1152 B64 transfers / 128 thr
      int idx = s * 128 + tid;
      __builtin_amdgcn_global_load_async_to_lds_b64(
          (gas1_v2i)(srcb + idx * 4), (gas3_v2i)(xrow + idx * 4), 0, 0);
    }
    asm volatile("s_wait_asynccnt 0x0" ::: "memory");
#else
#pragma unroll
    for (int s = 0; s < 9; ++s) {
      int idx = s * 128 + tid;
      u16x4 vv = ((const u16x4*)srcb)[idx];
      ((u16x4*)xrow)[idx] = vv;
      if (u < Rn - 1)
        __builtin_prefetch(((const u16x4*)srcb) + 1152 + idx, 0, 0);
    }
#endif
    __syncthreads();
#pragma unroll
    for (int vv = 0; vv < Rn; ++vv) {
#pragma unroll
      for (int ch = 0; ch < 2; ++ch) {
        int off = (j + vv) * Cn + ch * 32 + kb;
        v8bf lo = *(const v8bf*)(xrow + off);        // c = base+0..7
        v8bf hi = *(const v8bf*)(xrow + off + 16);   // c = base+16..23
        v16bf a = __builtin_shufflevector(lo, hi, 0, 1, 2, 3, 4, 5, 6, 7,
                                          8, 9, 10, 11, 12, 13, 14, 15);
        int kc = (u * 9 + vv) * 2 + ch;
        v16bf bfv = *(const v16bf*)(pf + ((size_t)kc * 32 + lane) * 16);
        acc = __builtin_amdgcn_wmma_f32_16x16x32_bf16(
            false, a, false, bfv, (short)0, acc, false, false);
      }
    }
  }

  // D: lane col n = filter, rows m = pixels -> e = 2*(rec - inputs), bf16
  int f = lane & 15;
  if (f < Fn) {
    int mb = (lane >> 4) * 8;
    size_t base = (((size_t)b * Fn + f) * Hn + i) * Wn + j0 + mb;
    v8bf ev;
#pragma unroll
    for (int r = 0; r < 8; ++r)
      ev[r] = (__bf16)(2.f * (acc[r] - inp[base + r]));
    *(v8bf*)(ebf + base) = ev;              // 16B aligned (j0+mb % 8 == 0)
  }
}

// =====================================================================
// Backward implicit GEMM + fused RMSProp.  Block = 20 waves = one (b,p).
// Zero-padded bf16 residual tile + k->offset table in LDS: hot loop is
// ds_load_b128 (table) + ds_load_u16 (e) + WMMA; no bounds checks/divides.
// Lanes update 8 consecutive NHWC channels (contiguous 32B) and also
// write the bf16 shadow of x consumed by the next forward pass.
// =====================================================================
__global__ __launch_bounds__(640)
void bwd_update(float* __restrict__ xw, float* __restrict__ a1,
                float* __restrict__ a2, __bf16* __restrict__ xbf,
                const __bf16* __restrict__ ebf, const __bf16* __restrict__ pb) {
  __shared__ __bf16 epad[4 * Rn * 96];      // 3456 bf16 (plane f=3 stays 0)
  __shared__ unsigned short tbl[256];       // k -> epad base offset

  const int tid  = threadIdx.x;
  const int lane = tid & 31;
  const int wave = tid >> 5;                // 0..19
  const int qs   = wave >> 2;               // 0..4 (q strip)
  const int cb   = wave & 3;                // channel block
  const int b    = blockIdx.x / Pn;
  const int p    = blockIdx.x - b * Pn;

  for (int idx = tid; idx < 4 * Rn * 96; idx += 640) epad[idx] = (__bf16)0.f;
  if (tid < 256) {
    int k = tid, base;
    if (k < KBWD) {
      int f = k / 81, rem = k - f * 81;
      int u = rem / 9, v = rem - u * 9;
      base = (f * Rn + u) * 96 + 8 - v;
    } else {
      base = 3 * Rn * 96 + 8;               // zero plane for K padding
    }
    tbl[k] = (unsigned short)base;
  }
  __syncthreads();

  const __bf16* eb = ebf + (size_t)b * Fn * Hn * Wn;
  for (int idx = tid; idx < Fn * Rn * Wn; idx += 640) {   // 1728 values
    int col = idx & 63;
    int fu  = idx >> 6;                     // 0..26
    int f = fu / 9, u = fu - f * 9;
    int ii = p - u;
    if (ii >= 0 && ii < Hn)
      epad[(f * Rn + u) * 96 + 8 + col] = eb[(f * Hn + ii) * Wn + col];
  }
  __syncthreads();

  const int q0 = qs * 16;
  const int n  = lane & 15;                 // B col = pixel in strip
  const int q  = q0 + n;

  v8f acc = {0.f, 0.f, 0.f, 0.f, 0.f, 0.f, 0.f, 0.f};

#pragma unroll
  for (int kc = 0; kc < KBWD_CH; ++kc) {
    int k0 = kc * 32 + (lane >> 4) * 16;    // 16 consecutive k per lane
    u16x8 t0 = *(const u16x8*)(tbl + k0);
    u16x8 t1 = *(const u16x8*)(tbl + k0 + 8);
    v16bf bfv;
#pragma unroll
    for (int t = 0; t < 8; ++t) {
      bfv[t]     = epad[(int)t0[t] + q];
      bfv[8 + t] = epad[(int)t1[t] + q];
    }
    v16bf af = *(const v16bf*)(pb + (((size_t)cb * KBWD_CH + kc) * 32 + lane) * 16);
    acc = __builtin_amdgcn_wmma_f32_16x16x32_bf16(
        false, af, false, bfv, (short)0, acc, false, false);
  }

  // D: lane col n = pixel, rows m = 8 consecutive channels (NHWC)
  if (q < Qn) {
    int mb = (lane >> 4) * 8;
    size_t base = (((size_t)b * Pn + p) * Qn + q) * Cn + cb * 16 + mb;
    v8bf xs;
#pragma unroll
    for (int r = 0; r < 8; ++r) {
      size_t idx = base + r;
      float xv  = xw[idx];
      float g   = acc[r] + GAMMA_ * xv * __frsqrt_rn(xv * xv + DIFF_EPS_);
      float a1n = RHO_ * a1[idx] + (1.f - RHO_) * g * g;
      float upd = LR_ * g * __frsqrt_rn(a1n + EPS_RMS);
      float a2n = MOM_ * a2[idx] - upd;
      float xn  = xv + MOM_ * a2n - upd;
      xw[idx]  = xn;
      a1[idx]  = a1n;
      a2[idx]  = a2n;
      xs[r] = (__bf16)xn;
    }
    *(v8bf*)(xbf + base) = xs;              // bf16 shadow, 16B aligned
  }
}

// NHWC workspace -> NCHW output (once, after the 16 iterations)
__global__ void transpose_out(const float* __restrict__ xw,
                              float* __restrict__ out) {
  size_t idx = (size_t)blockIdx.x * blockDim.x + threadIdx.x;
  const size_t total = (size_t)Bn * Cn * Pn * Qn;
  if (idx < total) {
    int q = (int)(idx % Qn);
    size_t r1 = idx / Qn;
    int pp = (int)(r1 % Pn);
    size_t r2 = r1 / Pn;
    int c = (int)(r2 % Cn);
    int b = (int)(r2 / Cn);
    out[idx] = xw[(((size_t)b * Pn + pp) * Qn + q) * Cn + c];
  }
}

// =====================================================================
// Host side: pack + init once, 16 fwd/bwd rounds, final transpose.
// ws: x(f32) | a1 | a2 | xbf(bf16) | ebf(bf16) | pf | pb   (~76 MB, L2-resident)
// =====================================================================
extern "C" void kernel_launch(void* const* d_in, const int* in_sizes, int n_in,
                              void* d_out, int out_size, void* d_ws, size_t ws_size,
                              hipStream_t stream) {
  const float* inp = (const float*)d_in[0];   // (16,3,64,64)
  const float* Wg  = (const float*)d_in[1];   // (3,64,9,9)

  const size_t nstate = (size_t)Bn * Cn * Pn * Qn;   // 5,308,416
  const size_t nres   = (size_t)Bn * Fn * Hn * Wn;   // 786,432

  float*  xw  = (float*)d_ws;                 // NHWC f32 master
  float*  a1  = xw + nstate;
  float*  a2  = a1 + nstate;
  __bf16* xbf = (__bf16*)(a2 + nstate);       // NHWC bf16 shadow
  __bf16* ebf = xbf + nstate;                 // NCHW bf16 residual
  __bf16* pf  = ebf + nres;                   // 32B aligned
  __bf16* pb  = pf + (size_t)KFWD_CH * 32 * 16;

  pack_weights<<<(KFWD_CH * 32 * 16 + 255) / 256, 256, 0, stream>>>(Wg, pf, pb);
  init_state<<<(int)((nstate + 255) / 256), 256, 0, stream>>>(xw, a1, a2, xbf,
                                                              (int)nstate);

  const int fwd_blocks = Bn * Hn;   // 1024 blocks x 128 (4 waves)
  const int bwd_blocks = Bn * Pn;   // 1152 blocks x 640 (20 waves)

  for (int it = 0; it < 16; ++it) {
    fwd_residual<<<fwd_blocks, 128, 0, stream>>>(xbf, inp, pf, ebf);
    bwd_update<<<bwd_blocks, 640, 0, stream>>>(xw, a1, a2, xbf, ebf, pb);
  }

  transpose_out<<<(int)((nstate + 255) / 256), 256, 0, stream>>>(xw, (float*)d_out);
}
